// Attention_18554258718902
// MI455X (gfx1250) — matrix operand order repaired and checked by reading, not compiled, not run on hardware
//
#include <hip/hip_runtime.h>

#define B_      2
#define N_      1024
#define DIM_    768
#define HEADS_  12
#define DH_     64
#define INNER_  768
#define QKV3_   2304

typedef __attribute__((ext_vector_type(16))) __bf16 bf16x16;
typedef __attribute__((ext_vector_type(8)))  __bf16 bf16x8;
typedef __attribute__((ext_vector_type(4)))  __bf16 bf16x4;
typedef __attribute__((ext_vector_type(4)))  float  f32x4;
typedef __attribute__((ext_vector_type(8)))  float  f32x8;

union BF16x16u { bf16x16 v; bf16x8 h[2]; __bf16 e[16]; };
union BF16x4u  { bf16x4 v;  __bf16 e[4]; };
union F32x4u   { f32x4 v;   float  e[4]; };

__device__ __forceinline__ f32x8 wmma_bf16(bf16x16 a, bf16x16 b, f32x8 c) {
  return __builtin_amdgcn_wmma_f32_16x16x32_bf16(false, a, false, b, (short)0, c, false, false);
}

// A-fragment (16x32 bf16, M x K), A row-major with leading dim lda.
// lane(lo,hi): VGPR p<4 -> K = hi*8+2p, p>=4 -> K = 16+hi*8+2(p-4)
// => two contiguous 16B loads at k = kc+hi*8 and kc+16+hi*8.
__device__ __forceinline__ bf16x16 load_a_frag(const __bf16* A, int lda, int row16, int kc, int lane) {
  const int lo = lane & 15, hi = lane >> 4;
  const __bf16* p = A + (size_t)(row16 + lo) * lda + kc + hi * 8;
  BF16x16u r;
  r.h[0] = *(const bf16x8*)(p);
  r.h[1] = *(const bf16x8*)(p + 16);
  return r.v;
}

// B-fragment (32x16 bf16, K x N) for computing A @ S^T where S is row-major [n][k].
// Lane lo = column n; same K order as A: elems 0-7 K=kc+hi*8+0..7, elems 8-15 K=kc+16+hi*8+0..7
// => two contiguous 16B loads per lane.
__device__ __forceinline__ bf16x16 load_b_frag(const __bf16* S, int lds, int col16, int kc, int lane) {
  const int lo = lane & 15, hi = lane >> 4;
  const __bf16* p = S + (size_t)(col16 + lo) * lds + kc + hi * 8;
  BF16x16u r;
  r.h[0] = *(const bf16x8*)(p);
  r.h[1] = *(const bf16x8*)(p + 16);
  return r.v;
}

// ---------------- Kernel 0: f32 -> bf16 bulk convert (weights, one-shot) ----------------
__global__ void k_cvt(const float* __restrict__ src, __bf16* __restrict__ dst, int n4) {
  const int i = blockIdx.x * blockDim.x + threadIdx.x;
  if (i < n4) {
    F32x4u u; u.v = *(const f32x4*)(src + (size_t)i * 4);
    BF16x4u o;
#pragma unroll
    for (int x = 0; x < 4; ++x) o.e[x] = (__bf16)u.e[x];
    *(bf16x4*)(dst + (size_t)i * 4) = o.v;
  }
}

// ---------------- Kernel 1: LayerNorm -> bf16 ----------------
__global__ void k_ln(const float* __restrict__ x, const float* __restrict__ g,
                     const float* __restrict__ bta, __bf16* __restrict__ xn) {
  __shared__ float red[256];
  const int row = blockIdx.x;                 // 0..2047
  const float* xr = x + (size_t)row * DIM_;
  float s = 0.f, s2 = 0.f;
  for (int i = threadIdx.x; i < DIM_; i += 256) { float v = xr[i]; s += v; s2 += v * v; }
  red[threadIdx.x] = s; __syncthreads();
  for (int o = 128; o > 0; o >>= 1) { if (threadIdx.x < o) red[threadIdx.x] += red[threadIdx.x + o]; __syncthreads(); }
  const float mean = red[0] * (1.0f / DIM_);
  __syncthreads();
  red[threadIdx.x] = s2; __syncthreads();
  for (int o = 128; o > 0; o >>= 1) { if (threadIdx.x < o) red[threadIdx.x] += red[threadIdx.x + o]; __syncthreads(); }
  const float var = red[0] * (1.0f / DIM_) - mean * mean;
  const float rs = rsqrtf(var + 1e-5f);
  for (int i = threadIdx.x; i < DIM_; i += 256) {
    float v = (xr[i] - mean) * rs * g[i] + bta[i];
    xn[(size_t)row * DIM_ + i] = (__bf16)v;
  }
}

// ---------------- Kernel 2: QKV GEMM (xn @ Wqkv^T), scatter q/k/vT ----------------
// 128 threads = 4 waves (2x2); each wave: 16 rows x 32 cols (2 tiles, shared A frag).
__global__ void k_qkv(const __bf16* __restrict__ xn, const __bf16* __restrict__ wqkv,
                      __bf16* __restrict__ q, __bf16* __restrict__ kk, __bf16* __restrict__ vT) {
  const int lane = threadIdx.x & 31, wave = threadIdx.x >> 5;
  const int rowBase = blockIdx.y * 32 + (wave >> 1) * 16;
  const int colBase = blockIdx.x * 64 + (wave & 1) * 32;
  f32x8 acc0 = {}, acc1 = {};
  for (int kc = 0; kc < DIM_; kc += 32) {
    bf16x16 a  = load_a_frag(xn, DIM_, rowBase, kc, lane);
    bf16x16 b0 = load_b_frag(wqkv, DIM_, colBase, kc, lane);
    bf16x16 b1 = load_b_frag(wqkv, DIM_, colBase + 16, kc, lane);
    acc0 = wmma_bf16(a, b0, acc0);
    acc1 = wmma_bf16(a, b1, acc1);
  }
  const int lo = lane & 15, hi = lane >> 4;
#pragma unroll
  for (int ct = 0; ct < 2; ++ct) {
    const f32x8 acc = ct ? acc1 : acc0;
    const int o = colBase + ct * 16 + lo;        // output feature 0..2303 (fixed per lane)
    const int part = o / INNER_;                 // 0=q 1=k 2=v
    const int oo = o % INNER_;
    const int h = oo / DH_, dd = oo % DH_;
#pragma unroll
    for (int r = 0; r < 8; ++r) {
      const int grow = rowBase + hi * 8 + r;     // 0..2047
      const int bb = grow >> 10, n = grow & 1023;
      const __bf16 val = (__bf16)acc[r];
      const size_t bh = (size_t)(bb * HEADS_ + h);
      if (part == 0)      q [(bh * N_ + n) * DH_ + dd] = val;
      else if (part == 1) kk[(bh * N_ + n) * DH_ + dd] = val;
      else                vT[(bh * DH_ + dd) * N_ + n] = val;   // store V transposed
    }
  }
}

// ---------------- Kernel 3: dots = scale*QK^T, row softmax -> bf16 attn ----------------
// 256 threads = 8 waves. Block owns a 16-row strip x all 1024 cols of one (b,h).
// Wave w owns columns [w*128, w*128+128) = 8 WMMA tiles.
__global__ void k_attn(const __bf16* __restrict__ q, const __bf16* __restrict__ kk,
                       __bf16* __restrict__ attn) {
  __shared__ float smax[8][16];
  __shared__ float ssum[8][16];
  const int lane = threadIdx.x & 31, wave = threadIdx.x >> 5;
  const int bh = blockIdx.x >> 6;
  const int rowBase = (blockIdx.x & 63) * 16;
  const __bf16* Q = q  + (size_t)bh * N_ * DH_;
  const __bf16* K = kk + (size_t)bh * N_ * DH_;
  f32x8 acc[8];
#pragma unroll
  for (int t = 0; t < 8; ++t) { f32x8 z = {}; acc[t] = z; }
#pragma unroll
  for (int kc = 0; kc < DH_; kc += 32) {
    bf16x16 a = load_a_frag(Q, DH_, rowBase, kc, lane);
#pragma unroll
    for (int t = 0; t < 8; ++t) {
      bf16x16 b = load_b_frag(K, DH_, wave * 128 + t * 16, kc, lane);
      acc[t] = wmma_bf16(a, b, acc[t]);
    }
  }
  const int lo = lane & 15, hi = lane >> 4;
  const float SCALE = 0.125f;                      // 64^-0.5
#pragma unroll
  for (int t = 0; t < 8; ++t)
#pragma unroll
    for (int r = 0; r < 8; ++r) acc[t][r] *= SCALE;

  float rmax[8], rsum[8];
#pragma unroll
  for (int r = 0; r < 8; ++r) {
    float m = -3.4e38f;
#pragma unroll
    for (int t = 0; t < 8; ++t) m = fmaxf(m, acc[t][r]);
    for (int off = 1; off < 16; off <<= 1) m = fmaxf(m, __shfl_xor(m, off, 32));
    rmax[r] = m;
  }
  if (lo == 0) {
#pragma unroll
    for (int r = 0; r < 8; ++r) smax[wave][hi * 8 + r] = rmax[r];
  }
  __syncthreads();
#pragma unroll
  for (int r = 0; r < 8; ++r) {
    float m = -3.4e38f;
#pragma unroll
    for (int w2 = 0; w2 < 8; ++w2) m = fmaxf(m, smax[w2][hi * 8 + r]);
    rmax[r] = m;
  }
#pragma unroll
  for (int r = 0; r < 8; ++r) {
    float s = 0.f;
#pragma unroll
    for (int t = 0; t < 8; ++t) { float e = __expf(acc[t][r] - rmax[r]); acc[t][r] = e; s += e; }
    for (int off = 1; off < 16; off <<= 1) s += __shfl_xor(s, off, 32);
    rsum[r] = s;
  }
  if (lo == 0) {
#pragma unroll
    for (int r = 0; r < 8; ++r) ssum[wave][hi * 8 + r] = rsum[r];
  }
  __syncthreads();
#pragma unroll
  for (int r = 0; r < 8; ++r) {
    float s = 0.f;
#pragma unroll
    for (int w2 = 0; w2 < 8; ++w2) s += ssum[w2][hi * 8 + r];
    rsum[r] = 1.0f / s;
  }
  __bf16* dst = attn + (size_t)bh * N_ * N_;
#pragma unroll
  for (int t = 0; t < 8; ++t)
#pragma unroll
    for (int r = 0; r < 8; ++r)
      dst[(size_t)(rowBase + hi * 8 + r) * N_ + wave * 128 + t * 16 + lo] = (__bf16)(acc[t][r] * rsum[r]);
}

// ---------------- Kernel 4: M = -0.5 * theta @ gnn_w (exact: softmax rows sum to 1) ----
__global__ void k_mix(const float* __restrict__ theta, const float* __restrict__ gw,
                      float* __restrict__ M) {
  if (threadIdx.x < 144) {
    const int i = threadIdx.x / 12, j = threadIdx.x % 12;
    float s = 0.f;
    for (int t = 0; t < 12; ++t) s += theta[i * 12 + t] * gw[t * 12 + j];
    M[i * 12 + j] = -0.5f * s;
  }
}

// ---------------- Kernel 5: in-place comb = attn + relu(M @_heads attn) ----------------
__global__ void k_comb(__bf16* __restrict__ attn, const float* __restrict__ M) {
  __shared__ float Ms[144];
  if (threadIdx.x < 144) Ms[threadIdx.x] = M[threadIdx.x];
  __syncthreads();
  const size_t idx = (size_t)blockIdx.x * blockDim.x + threadIdx.x;   // 524288 threads
  const size_t bb  = idx >> 18;
  const size_t pos = (idx & 262143) << 2;                              // 4 elems/thread
  __bf16* base = attn + bb * (size_t)HEADS_ * N_ * N_ + pos;
  const size_t HS = (size_t)N_ * N_;
  float a[12][4];
#pragma unroll
  for (int h = 0; h < 12; ++h) {
    BF16x4u u; u.v = *(const bf16x4*)(base + h * HS);
#pragma unroll
    for (int x = 0; x < 4; ++x) a[h][x] = (float)u.e[x];
  }
#pragma unroll
  for (int i = 0; i < 12; ++i) {
    float s0 = 0.f, s1 = 0.f, s2 = 0.f, s3 = 0.f;
#pragma unroll
    for (int j = 0; j < 12; ++j) {
      const float m = Ms[i * 12 + j];
      s0 += m * a[j][0]; s1 += m * a[j][1]; s2 += m * a[j][2]; s3 += m * a[j][3];
    }
    BF16x4u o;
    o.e[0] = (__bf16)(a[i][0] + fmaxf(s0, 0.f));
    o.e[1] = (__bf16)(a[i][1] + fmaxf(s1, 0.f));
    o.e[2] = (__bf16)(a[i][2] + fmaxf(s2, 0.f));
    o.e[3] = (__bf16)(a[i][3] + fmaxf(s3, 0.f));
    *(bf16x4*)(base + i * HS) = o.v;
  }
}

// ---------------- Kernel 6: out_head = comb @ V  (via V^T), store [B,N,768] bf16 -------
// 256 threads = 8 waves (2 row-groups x 4 col-tiles); each wave: 2 row tiles, shared B.
__global__ void k_av(const __bf16* __restrict__ comb, const __bf16* __restrict__ vT,
                     __bf16* __restrict__ Xh) {
  const int lane = threadIdx.x & 31, wave = threadIdx.x >> 5;
  const int bh = blockIdx.x >> 4;                       // 24 (b,h) x 16 row blocks
  const int rowBase = (blockIdx.x & 15) * 64 + (wave >> 2) * 32;
  const int colBase = (wave & 3) * 16;
  const __bf16* A = comb + (size_t)bh * N_ * N_;
  const __bf16* S = vT   + (size_t)bh * DH_ * N_;
  f32x8 acc0 = {}, acc1 = {};
  for (int kc = 0; kc < N_; kc += 32) {
    bf16x16 a0 = load_a_frag(A, N_, rowBase,      kc, lane);
    bf16x16 a1 = load_a_frag(A, N_, rowBase + 16, kc, lane);
    bf16x16 b  = load_b_frag(S, N_, colBase, kc, lane);
    acc0 = wmma_bf16(a0, b, acc0);
    acc1 = wmma_bf16(a1, b, acc1);
  }
  const int lo = lane & 15, hi = lane >> 4;
  const int bb = bh / HEADS_, h = bh % HEADS_;
  const int d = colBase + lo;
#pragma unroll
  for (int rt = 0; rt < 2; ++rt) {
    const f32x8 acc = rt ? acc1 : acc0;
#pragma unroll
    for (int r = 0; r < 8; ++r) {
      const int n = rowBase + rt * 16 + hi * 8 + r;
      Xh[((size_t)(bb * N_ + n)) * INNER_ + h * DH_ + d] = (__bf16)acc[r];
    }
  }
}

// ---------------- Kernel 7: out = Xh @ Wout^T + b_out (f32 result) ----------------
// 128 threads = 4 waves (2x2); each wave: 16 rows x 32 cols (2 tiles, shared A frag).
__global__ void k_out(const __bf16* __restrict__ Xh, const __bf16* __restrict__ wout,
                      const float* __restrict__ bout, float* __restrict__ out) {
  const int lane = threadIdx.x & 31, wave = threadIdx.x >> 5;
  const int rowBase = blockIdx.y * 32 + (wave >> 1) * 16;
  const int colBase = blockIdx.x * 64 + (wave & 1) * 32;
  f32x8 acc0 = {}, acc1 = {};
  for (int kc = 0; kc < INNER_; kc += 32) {
    bf16x16 a  = load_a_frag(Xh, INNER_, rowBase, kc, lane);
    bf16x16 b0 = load_b_frag(wout, INNER_, colBase, kc, lane);
    bf16x16 b1 = load_b_frag(wout, INNER_, colBase + 16, kc, lane);
    acc0 = wmma_bf16(a, b0, acc0);
    acc1 = wmma_bf16(a, b1, acc1);
  }
  const int lo = lane & 15, hi = lane >> 4;
#pragma unroll
  for (int ct = 0; ct < 2; ++ct) {
    const f32x8 acc = ct ? acc1 : acc0;
    const int o = colBase + ct * 16 + lo;
    const float bias = bout[o];
#pragma unroll
    for (int r = 0; r < 8; ++r)
      out[(size_t)(rowBase + hi * 8 + r) * DIM_ + o] = acc[r] + bias;
  }
}

extern "C" void kernel_launch(void* const* d_in, const int* in_sizes, int n_in,
                              void* d_out, int out_size, void* d_ws, size_t ws_size,
                              hipStream_t stream) {
  (void)in_sizes; (void)n_in; (void)out_size; (void)ws_size;
  const float* x     = (const float*)d_in[0];
  const float* ln_g  = (const float*)d_in[1];
  const float* ln_b  = (const float*)d_in[2];
  const float* w_qkv = (const float*)d_in[3];
  const float* w_out = (const float*)d_in[4];
  const float* b_out = (const float*)d_in[5];
  const float* theta = (const float*)d_in[6];
  const float* gnn_w = (const float*)d_in[7];
  float* out = (float*)d_out;

  char* ws = (char*)d_ws;
  const size_t SZ_QKVH = (size_t)B_ * HEADS_ * N_ * DH_ * sizeof(__bf16);   // 3 MB each
  const size_t SZ_XN   = (size_t)B_ * N_ * DIM_ * sizeof(__bf16);           // 3 MB
  const size_t SZ_ATTN = (size_t)B_ * HEADS_ * N_ * N_ * sizeof(__bf16);    // 50.3 MB
  __bf16* qb    = (__bf16*)(ws);
  __bf16* kb    = (__bf16*)(ws + SZ_QKVH);
  __bf16* vTb   = (__bf16*)(ws + 2 * SZ_QKVH);
  __bf16* xnb   = (__bf16*)(ws + 3 * SZ_QKVH);
  __bf16* Xhb   = (__bf16*)(ws + 3 * SZ_QKVH + SZ_XN);
  float*  Mb    = (float*) (ws + 3 * SZ_QKVH + 2 * SZ_XN);
  __bf16* attn  = (__bf16*)(ws + 3 * SZ_QKVH + 2 * SZ_XN + 1024);
  __bf16* wqkvb = (__bf16*)(ws + 3 * SZ_QKVH + 2 * SZ_XN + 1024 + SZ_ATTN);
  __bf16* woutb = (__bf16*)(ws + 3 * SZ_QKVH + 2 * SZ_XN + 1024 + SZ_ATTN
                              + (size_t)QKV3_ * DIM_ * sizeof(__bf16));

  // 0) one-shot weight conversions (L2-resident bf16 copies)
  k_cvt<<<(QKV3_ * DIM_ / 4 + 255) / 256, 256, 0, stream>>>(w_qkv, wqkvb, QKV3_ * DIM_ / 4);
  k_cvt<<<(INNER_ * DIM_ / 4 + 255) / 256, 256, 0, stream>>>(w_out, woutb, INNER_ * DIM_ / 4);
  // 1) LayerNorm
  k_ln<<<B_ * N_, 256, 0, stream>>>(x, ln_g, ln_b, xnb);
  // 2) tiny head-mix matrix (independent of attn; run early)
  k_mix<<<1, 192, 0, stream>>>(theta, gnn_w, Mb);
  // 3) QKV projection
  k_qkv<<<dim3(QKV3_ / 64, (B_ * N_) / 32), 128, 0, stream>>>(xnb, wqkvb, qb, kb, vTb);
  // 4) QK^T + softmax
  k_attn<<<B_ * HEADS_ * (N_ / 16), 256, 0, stream>>>(qb, kb, attn);
  // 5) in-place head mixing + relu residual
  k_comb<<<2048, 256, 0, stream>>>(attn, Mb);
  // 6) (attn + H) @ V
  k_av<<<B_ * HEADS_ * (N_ / 64), 256, 0, stream>>>(attn, vTb, Xhb);
  // 7) output projection + bias
  k_out<<<dim3(INNER_ / 64, (B_ * N_) / 32), 128, 0, stream>>>(Xhb, woutb, b_out, out);
}
